// Evoformer_20753281974323
// MI455X (gfx1250) — compile-verified, matrix operand order checked by reading, not run on hardware
//
#include <hip/hip_runtime.h>
#include <hip/hip_bf16.h>

// ---------------------------------------------------------------------------
// AlphaFold2 Evoformer MSA row-attention with pair bias, for gfx1250 (CDNA5).
// GEMMs and attention: V_WMMA_F32_16X16X32_F16 fed from LDS tiles filled by
// double-buffered GLOBAL_LOAD_ASYNC_TO_LDS_B128 (ASYNCcnt-tracked) copies.
// ---------------------------------------------------------------------------

typedef __attribute__((ext_vector_type(16))) _Float16 v16h;
typedef __attribute__((ext_vector_type(8)))  float    v8f;

#define C_S   256
#define C_Z   128
#define NH    8
#define DH    32
#define TDIM  256
#define SEQ   128
#define NRES  256
#define SN    (SEQ * NRES)           // 32768 rows
#define SCALE 0.17677669529663687f   // 1/sqrt(32)

// ---------------------------------------------------------------------------
// Async 16-byte copy global -> LDS (per-lane addresses), CDNA5 async path.
// VDST = LDS byte offset VGPR, VADDR = 64-bit global address VGPR pair.
// ---------------------------------------------------------------------------
__device__ __forceinline__ void async_copy16(unsigned int lds_byte_off,
                                             const _Float16* gptr) {
  unsigned long long ga = (unsigned long long)(uintptr_t)gptr;
  asm volatile("global_load_async_to_lds_b128 %0, %1, off"
               :: "v"(lds_byte_off), "v"(ga) : "memory");
}

// ---------------------------------------------------------------------------
// WMMA fragment loader for row-major f16 (16 rows x 32 K), wave32 layout:
//   lane: m = lane%16 (row), hf = lane/16
//   elements e=0..7  -> k = hf*8 + e ; e=8..15 -> k = 16 + hf*8 + (e-8)
// Serves B-fragments too when the source is stored [n][k] (B^T row-major).
// ---------------------------------------------------------------------------
__device__ __forceinline__ v16h load_frag16(const _Float16* base, int ld) {
  const int lane = threadIdx.x & 31;
  const int m = lane & 15;
  const int hf = lane >> 4;
  const _Float16* p = base + (size_t)m * ld + hf * 8;
  union { v16h v; uint4 q[2]; } u;
  u.q[0] = *(const uint4*)(p);
  u.q[1] = *(const uint4*)(p + 16);
  return u.v;
}

// ---------------------------------------------------------------------------
// Kernel 1: LayerNorm of MSA -> f16. One wave per 256-wide row.
// ---------------------------------------------------------------------------
__global__ void ln_msa_kernel(const float* __restrict__ x,
                              const float* __restrict__ w,
                              const float* __restrict__ b,
                              _Float16* __restrict__ out) {
  const int wid = threadIdx.x >> 5, lane = threadIdx.x & 31;
  const size_t row = (size_t)blockIdx.x * 8 + wid;
  const float* xr = x + row * C_S;
  float v[8], sum = 0.f, sq = 0.f;
#pragma unroll
  for (int t = 0; t < 8; ++t) {
    const float val = xr[lane + 32 * t];
    v[t] = val; sum += val; sq += val * val;
  }
#pragma unroll
  for (int off = 16; off; off >>= 1) {
    sum += __shfl_xor(sum, off, 32);
    sq  += __shfl_xor(sq,  off, 32);
  }
  const float mean = sum * (1.f / C_S);
  const float var  = sq * (1.f / C_S) - mean * mean;
  const float rs   = rsqrtf(var + 1e-5f);
#pragma unroll
  for (int t = 0; t < 8; ++t) {
    const int c = lane + 32 * t;
    out[row * C_S + c] = (_Float16)((v[t] - mean) * rs * w[c] + b[c]);
  }
}

// ---------------------------------------------------------------------------
// Kernel 2: transpose fp32 weight [K][N] -> f16 [N][K] (contiguous B-frags).
// ---------------------------------------------------------------------------
__global__ void transpose_f32_to_f16(const float* __restrict__ W,
                                     _Float16* __restrict__ Wt,
                                     int K, int N) {
  const int n = blockIdx.x * 16 + threadIdx.x;
  const int k = blockIdx.y * 16 + threadIdx.y;
  if (n < N && k < K) Wt[(size_t)n * K + k] = (_Float16)W[(size_t)k * N + n];
}

// ---------------------------------------------------------------------------
// Kernel 3: WMMA GEMM  C[M,N] = A[M,K] @ Bt[N,K]^T with double-buffered
// async-to-LDS staging. Block = 256 threads = 8 waves; block tile 128x64,
// wave tile 16x64, K-step 32. Each stage: 12 KB staged by 3 async B128
// instructions per wave; s_wait_asynccnt 3 overlaps stage k+1 copy with
// stage k WMMA.
// mode 0: f16 store; mode 1: fp32 sigmoid(acc+bias[n]); mode 2: fp32
// acc+bias[n]; mode 3: f16 store transposed per-s (V -> Vt[s][d][j]).
// ---------------------------------------------------------------------------
__global__ void gemm_f16_kernel(const _Float16* __restrict__ A,
                                const _Float16* __restrict__ Bt,
                                void* __restrict__ Cout,
                                const float* __restrict__ bias_vec,
                                int M, int N, int K, int mode) {
  __shared__ __align__(16) _Float16 As[2][128][32];
  __shared__ __align__(16) _Float16 Bs[2][64][32];
  const int wid = threadIdx.x >> 5, lane = threadIdx.x & 31;
  const int tid = threadIdx.x;
  const int row0 = blockIdx.x * 128;
  const int col0 = blockIdx.y * 64;
  const int nk = K >> 5;

  // Cooperative stage: A = 512 x 16B chunks, B = 256 x 16B chunks;
  // chunk c -> row c/4, half-offset (c%4)*8. 3 async instrs per thread.
  auto stage = [&](int kt, int buf) {
#pragma unroll
    for (int c = 0; c < 2; ++c) {
      const int chunk = tid + 256 * c;
      const int r = chunk >> 2, o = (chunk & 3) * 8;
      async_copy16((unsigned int)(uintptr_t)&As[buf][r][o],
                   A + (size_t)(row0 + r) * K + kt * 32 + o);
    }
    {
      const int r = tid >> 2, o = (tid & 3) * 8;
      async_copy16((unsigned int)(uintptr_t)&Bs[buf][r][o],
                   Bt + (size_t)(col0 + r) * K + kt * 32 + o);
    }
  };

  v8f acc[4] = {};
  stage(0, 0);
  for (int kt = 0; kt < nk; ++kt) {
    const int buf = kt & 1;
    if (kt + 1 < nk) {
      stage(kt + 1, buf ^ 1);                          // prefetch next stage
      asm volatile("s_wait_asynccnt 3" ::: "memory");  // stage kt complete
    } else {
      asm volatile("s_wait_asynccnt 0" ::: "memory");
    }
    __syncthreads();
    const v16h a = load_frag16(&As[buf][wid * 16][0], 32);
#pragma unroll
    for (int t = 0; t < 4; ++t) {
      const v16h b = load_frag16(&Bs[buf][t * 16][0], 32);
      acc[t] = __builtin_amdgcn_wmma_f32_16x16x32_f16(
          false, a, false, b, (short)0, acc[t], false, false);
    }
    __syncthreads();   // all waves done reading before buffer is re-filled
  }

  const int n16 = lane & 15, hf = lane >> 4;
#pragma unroll
  for (int t = 0; t < 4; ++t) {
#pragma unroll
    for (int r = 0; r < 8; ++r) {
      const int row = row0 + wid * 16 + r + 8 * hf;
      const int col = col0 + t * 16 + n16;
      const size_t idx = (size_t)row * N + col;
      float v = acc[t][r];
      if (mode == 0) {
        ((_Float16*)Cout)[idx] = (_Float16)v;
      } else if (mode == 1) {
        v += bias_vec[col];
        ((float*)Cout)[idx] = 1.f / (1.f + __expf(-v));
      } else if (mode == 2) {
        ((float*)Cout)[idx] = v + bias_vec[col];
      } else {  // mode 3: V transposed per-s -> Vt[s][col][j]
        const int srow = row >> 8, j = row & 255;
        ((_Float16*)Cout)[((size_t)srow * 256 + col) * 256 + j] = (_Float16)v;
      }
    }
  }
}

// ---------------------------------------------------------------------------
// Kernel 4: pair LayerNorm + projection to per-head bias [H][N*N] (fp32).
// ---------------------------------------------------------------------------
__global__ void pair_bias_kernel(const float* __restrict__ pair,
                                 const float* __restrict__ w,
                                 const float* __restrict__ b,
                                 const float* __restrict__ Wp,
                                 float* __restrict__ biasP) {
  const int wid = threadIdx.x >> 5, lane = threadIdx.x & 31;
  const size_t row = (size_t)blockIdx.x * 8 + wid;   // row = i*NRES + j
  const float* xr = pair + row * C_Z;
  float v[4], sum = 0.f, sq = 0.f;
#pragma unroll
  for (int t = 0; t < 4; ++t) {
    const float val = xr[lane + 32 * t];
    v[t] = val; sum += val; sq += val * val;
  }
#pragma unroll
  for (int off = 16; off; off >>= 1) {
    sum += __shfl_xor(sum, off, 32);
    sq  += __shfl_xor(sq,  off, 32);
  }
  const float mean = sum * (1.f / C_Z);
  const float var  = sq * (1.f / C_Z) - mean * mean;
  const float rs   = rsqrtf(var + 1e-5f);
  float hb[NH] = {};
#pragma unroll
  for (int t = 0; t < 4; ++t) {
    const int c = lane + 32 * t;
    const float xn = (v[t] - mean) * rs * w[c] + b[c];
#pragma unroll
    for (int hh = 0; hh < NH; ++hh) hb[hh] += xn * Wp[c * NH + hh];
  }
#pragma unroll
  for (int hh = 0; hh < NH; ++hh) {
    float a = hb[hh];
#pragma unroll
    for (int off = 16; off; off >>= 1) a += __shfl_xor(a, off, 32);
    if (lane == 0) biasP[(size_t)hh * (NRES * NRES) + row] = a;
  }
}

// ---------------------------------------------------------------------------
// Kernel 5: flash-style attention per (s, h). Block = 4 waves, each owning
// 16 query rows; Q resident (D=32 = one WMMA K-step). K and Vt j-blocks of
// 32 are staged into double-buffered LDS with async copies and shared by all
// waves. fp32 online softmax in C-fragment layout; P re-shaped C->A layout
// through LDS.
// ---------------------------------------------------------------------------
__global__ void msa_attention_kernel(const _Float16* __restrict__ Qh,
                                     const _Float16* __restrict__ Kh,
                                     const _Float16* __restrict__ Vt,
                                     const float* __restrict__ Gf,
                                     const float* __restrict__ biasP,
                                     _Float16* __restrict__ AVh) {
  __shared__ __align__(16) _Float16 Plds[4][16][32];
  __shared__ __align__(16) _Float16 Ks[2][32][32];
  __shared__ __align__(16) _Float16 Vs[2][32][32];
  const int wid = threadIdx.x >> 5, lane = threadIdx.x & 31;
  const int tid = threadIdx.x;
  const int s = blockIdx.x >> 3, h = blockIdx.x & 7;
  const int i0 = blockIdx.y * 64 + wid * 16;
  const size_t base = (size_t)s * NRES * TDIM + h * DH;  // row i -> base + i*TDIM
  const _Float16* qb = Qh + base;
  const _Float16* kb = Kh + base;
  // Vt[s][d_total][j] : rows are d, columns are j (contiguous).
  const _Float16* vtb = Vt + (size_t)s * (TDIM * NRES) + (size_t)(h * DH) * NRES;
  const float* bias = biasP + (size_t)h * (NRES * NRES);
  const int n16 = lane & 15, hf = lane >> 4;

  // Stage one 32-wide j-block of K (32 j-rows x 32 d) and Vt (32 d-rows x 32 j).
  // 128 chunks of 16B each per matrix -> 2 async instrs per thread.
  auto stageKV = [&](int j0s, int buf) {
    const int r = tid >> 2, o = (tid & 3) * 8;
    async_copy16((unsigned int)(uintptr_t)&Ks[buf][r][o],
                 kb + (size_t)(j0s + r) * TDIM + o);
    async_copy16((unsigned int)(uintptr_t)&Vs[buf][r][o],
                 vtb + (size_t)r * NRES + j0s + o);
  };

  const v16h qa = load_frag16(qb + (size_t)i0 * TDIM, TDIM);

  float mrow[8], lrow[8];
  v8f o0 = {}, o1 = {};
#pragma unroll
  for (int r = 0; r < 8; ++r) { mrow[r] = -1e30f; lrow[r] = 0.f; }

  stageKV(0, 0);
  for (int j0 = 0; j0 < NRES; j0 += 32) {
    const int buf = (j0 >> 5) & 1;
    if (j0 + 32 < NRES) {
      stageKV(j0 + 32, buf ^ 1);                       // prefetch next j-block
      asm volatile("s_wait_asynccnt 2" ::: "memory");  // current block complete
      __builtin_prefetch(bias + (size_t)i0 * NRES + j0 + 32, 0, 0);
    } else {
      asm volatile("s_wait_asynccnt 0" ::: "memory");
    }
    __syncthreads();

    const v16h kf0 = load_frag16(&Ks[buf][0][0], 32);
    const v16h kf1 = load_frag16(&Ks[buf][16][0], 32);
    const v8f z = {};
    v8f s0 = __builtin_amdgcn_wmma_f32_16x16x32_f16(false, qa, false, kf0,
                                                    (short)0, z, false, false);
    v8f s1 = __builtin_amdgcn_wmma_f32_16x16x32_f16(false, qa, false, kf1,
                                                    (short)0, z, false, false);
#pragma unroll
    for (int r = 0; r < 8; ++r) {
      const int i = i0 + r + 8 * hf;
      const float v0 = s0[r] * SCALE + bias[(size_t)i * NRES + j0 + n16];
      const float v1 = s1[r] * SCALE + bias[(size_t)i * NRES + j0 + 16 + n16];
      float mx = fmaxf(v0, v1);
#pragma unroll
      for (int off = 8; off; off >>= 1) mx = fmaxf(mx, __shfl_xor(mx, off, 32));
      const float mnew  = fmaxf(mrow[r], mx);
      const float alpha = __expf(mrow[r] - mnew);
      const float p0 = __expf(v0 - mnew);
      const float p1 = __expf(v1 - mnew);
      float rsum = p0 + p1;
#pragma unroll
      for (int off = 8; off; off >>= 1) rsum += __shfl_xor(rsum, off, 32);
      lrow[r] = lrow[r] * alpha + rsum;
      mrow[r] = mnew;
      o0[r] *= alpha;
      o1[r] *= alpha;
      Plds[wid][r + 8 * hf][n16]      = (_Float16)p0;
      Plds[wid][r + 8 * hf][n16 + 16] = (_Float16)p1;
    }
    __asm__ volatile("s_wait_dscnt 0" ::: "memory");  // LDS RAW fence (wave-local)
    const v16h pa  = load_frag16(&Plds[wid][0][0], 32);
    const v16h vb0 = load_frag16(&Vs[buf][0][0], 32);    // d 0..15
    const v16h vb1 = load_frag16(&Vs[buf][16][0], 32);   // d 16..31
    o0 = __builtin_amdgcn_wmma_f32_16x16x32_f16(false, pa, false, vb0,
                                                (short)0, o0, false, false);
    o1 = __builtin_amdgcn_wmma_f32_16x16x32_f16(false, pa, false, vb1,
                                                (short)0, o1, false, false);
    __syncthreads();   // all waves done with Ks/Vs[buf] before re-fill
  }

  // Normalize, gate, emit f16 for the output projection.
#pragma unroll
  for (int r = 0; r < 8; ++r) {
    const int i = i0 + r + 8 * hf;
    const size_t row = base + (size_t)i * TDIM;
    const float inv = 1.f / lrow[r];
    const float g0 = Gf[row + n16];
    const float g1 = Gf[row + 16 + n16];
    AVh[row + n16]      = (_Float16)(o0[r] * inv * g0);
    AVh[row + 16 + n16] = (_Float16)(o1[r] * inv * g1);
  }
}

// ---------------------------------------------------------------------------
// Host-side orchestration.
// ---------------------------------------------------------------------------
extern "C" void kernel_launch(void* const* d_in, const int* in_sizes, int n_in,
                              void* d_out, int out_size, void* d_ws, size_t ws_size,
                              hipStream_t stream) {
  (void)in_sizes; (void)n_in; (void)out_size; (void)ws_size;

  const float* msa       = (const float*)d_in[0];
  const float* pair      = (const float*)d_in[1];
  const float* ln_msa_w  = (const float*)d_in[2];
  const float* ln_msa_b  = (const float*)d_in[3];
  const float* ln_pair_w = (const float*)d_in[4];
  const float* ln_pair_b = (const float*)d_in[5];
  const float* Wq        = (const float*)d_in[6];
  const float* Wk        = (const float*)d_in[7];
  const float* Wv        = (const float*)d_in[8];
  const float* Wpair     = (const float*)d_in[9];
  const float* Wg        = (const float*)d_in[10];
  const float* bg        = (const float*)d_in[11];
  const float* Wo        = (const float*)d_in[12];
  const float* bo        = (const float*)d_in[13];
  float* out = (float*)d_out;

  // Workspace carve-up (256B aligned slices).
  char* ws = (char*)d_ws;
  auto alloc = [&](size_t bytes) -> char* {
    char* p = ws;
    ws += (bytes + 255) & ~(size_t)255;
    return p;
  };
  _Float16* msa_h = (_Float16*)alloc((size_t)SN * C_S * 2);
  _Float16* WqT   = (_Float16*)alloc((size_t)C_S * TDIM * 2);
  _Float16* WkT   = (_Float16*)alloc((size_t)C_S * TDIM * 2);
  _Float16* WvT   = (_Float16*)alloc((size_t)C_S * TDIM * 2);
  _Float16* WgT   = (_Float16*)alloc((size_t)C_S * TDIM * 2);
  _Float16* WoT   = (_Float16*)alloc((size_t)TDIM * C_S * 2);
  _Float16* Qh    = (_Float16*)alloc((size_t)SN * TDIM * 2);
  _Float16* Kh    = (_Float16*)alloc((size_t)SN * TDIM * 2);
  _Float16* Vt    = (_Float16*)alloc((size_t)SN * TDIM * 2);   // [S][256 d][256 j]
  float*    Gf    = (float*)   alloc((size_t)SN * TDIM * 4);
  float*    biasP = (float*)   alloc((size_t)NH * NRES * NRES * 4);
  _Float16* AVh   = (_Float16*)alloc((size_t)SN * TDIM * 2);

  // 1. MSA LayerNorm -> f16.
  ln_msa_kernel<<<SN / 8, 256, 0, stream>>>(msa, ln_msa_w, ln_msa_b, msa_h);

  // 2. Weight transpose+convert (B^T layout for WMMA B fragments).
  dim3 tb(16, 16), tg(16, 16);
  transpose_f32_to_f16<<<tg, tb, 0, stream>>>(Wq, WqT, C_S, TDIM);
  transpose_f32_to_f16<<<tg, tb, 0, stream>>>(Wk, WkT, C_S, TDIM);
  transpose_f32_to_f16<<<tg, tb, 0, stream>>>(Wv, WvT, C_S, TDIM);
  transpose_f32_to_f16<<<tg, tb, 0, stream>>>(Wg, WgT, C_S, TDIM);
  transpose_f32_to_f16<<<tg, tb, 0, stream>>>(Wo, WoT, TDIM, C_S);

  // 3. Q/K/V/G projections (async-LDS WMMA GEMM, 128x64 block tiles).
  dim3 ggrid(SN / 128, TDIM / 64);
  gemm_f16_kernel<<<ggrid, 256, 0, stream>>>(msa_h, WqT, Qh, nullptr,
                                             SN, TDIM, C_S, 0);
  gemm_f16_kernel<<<ggrid, 256, 0, stream>>>(msa_h, WkT, Kh, nullptr,
                                             SN, TDIM, C_S, 0);
  gemm_f16_kernel<<<ggrid, 256, 0, stream>>>(msa_h, WvT, Vt, nullptr,
                                             SN, TDIM, C_S, 3);
  gemm_f16_kernel<<<ggrid, 256, 0, stream>>>(msa_h, WgT, Gf, bg,
                                             SN, TDIM, C_S, 1);

  // 4. Pair bias.
  pair_bias_kernel<<<NRES * NRES / 8, 256, 0, stream>>>(
      pair, ln_pair_w, ln_pair_b, Wpair, biasP);

  // 5. Attention (WMMA QK^T + online softmax + WMMA PV + gate).
  msa_attention_kernel<<<dim3(SEQ * NH, NRES / 64), 128, 0, stream>>>(
      Qh, Kh, Vt, Gf, biasP, AVh);

  // 6. Output projection (WMMA) + bias -> fp32 out.
  gemm_f16_kernel<<<dim3(SN / 128, C_S / 64), 256, 0, stream>>>(
      AVh, WoT, out, bo, SN, C_S, TDIM, 2);
}
